// Attention_67336497266780
// MI455X (gfx1250) — matrix operand order repaired and checked by reading, not compiled, not run on hardware
//
#include <hip/hip_runtime.h>

// ---------------- types ----------------
typedef __attribute__((ext_vector_type(16))) __bf16 v16bf;
typedef __attribute__((ext_vector_type(8)))  __bf16 bf16x8;
typedef __attribute__((ext_vector_type(2)))  __bf16 bf16x2;
typedef __attribute__((ext_vector_type(8)))  float  v8f;
typedef __attribute__((ext_vector_type(4)))  float  f32x4;
typedef __attribute__((ext_vector_type(4)))  unsigned int u32x4;
typedef __attribute__((ext_vector_type(8)))  int    i32x8;
typedef __attribute__((ext_vector_type(4)))  int    i32x4;

#define B_DIM 8
#define S_DIM 2048
#define E_DIM 1024
#define H_DIM 64

// ============================================================
// Kernel 1: projection  out[r,h] = (sum_e X[r,e]*W[e,h] + bias[h]) * scale
// X: [B*S, E] fp32, W: [E, H] fp32, out: [B*S, H] bf16
// Block = 128 threads = 4 waves; each wave does a 16x64 tile.
// W k-slab staged in LDS transposed + pre-converted to bf16 so each
// B fragment is two contiguous ds_load_b128 with no converts.
// ============================================================
__global__ __launch_bounds__(128) void proj_kernel(
    const float* __restrict__ X, const float* __restrict__ W,
    const float* __restrict__ bias, float scale, __bf16* __restrict__ out)
{
    __shared__ __align__(16) __bf16 wslabT[H_DIM * 32];  // [n][k] bf16, 4 KB

    const int tid   = threadIdx.x;
    const int wave  = tid >> 5;
    const int lane  = tid & 31;
    const int lhalf = lane >> 4;     // which 16-lane half
    const int lmod  = lane & 15;
    const long rowBase = (long)blockIdx.x * 64 + wave * 16;

    v8f acc[4] = {};                 // 4 N-tiles of 16 -> H=64

    for (int k0 = 0; k0 < E_DIM; k0 += 32) {
        __syncthreads();
        // stage W[k0..k0+32, :] transposed->bf16: pair (k, k+1) per b32 store
#pragma unroll
        for (int t = 0; t < 8; ++t) {
            int p  = tid + 128 * t;          // 0..1023
            int n  = p & 63;
            int kp = p >> 6;                 // 0..15 -> k = 2*kp
            float w0 = W[(long)(k0 + 2 * kp) * H_DIM + n];
            float w1 = W[(long)(k0 + 2 * kp + 1) * H_DIM + n];
            bf16x2 pk; pk[0] = (__bf16)w0; pk[1] = (__bf16)w1;
            *(bf16x2*)&wslabT[n * 32 + 2 * kp] = pk;
        }
        __syncthreads();

        // A fragment (16x32 bf16): row = rowBase+lmod,
        // K runs: [lhalf*8 .. +8) and [16+lhalf*8 .. +8)
        const float* xrow = X + (rowBase + lmod) * (long)E_DIM + k0 + lhalf * 8;
        f32x4 a0 = *(const f32x4*)(xrow + 0);
        f32x4 a1 = *(const f32x4*)(xrow + 4);
        f32x4 a2 = *(const f32x4*)(xrow + 16);
        f32x4 a3 = *(const f32x4*)(xrow + 20);
        v16bf afrag;
#pragma unroll
        for (int e = 0; e < 4; ++e) {
            afrag[e]      = (__bf16)a0[e];
            afrag[4 + e]  = (__bf16)a1[e];
            afrag[8 + e]  = (__bf16)a2[e];
            afrag[12 + e] = (__bf16)a3[e];
        }

#pragma unroll
        for (int nt = 0; nt < 4; ++nt) {
            // B fragment (32x16): col = nt*16+lmod, k runs [lhalf*8..+8) and [16+lhalf*8..+8)
            const __bf16* wc = &wslabT[(nt * 16 + lmod) * 32 + lhalf * 8];
            bf16x8 b0 = *(const bf16x8*)(wc);
            bf16x8 b1 = *(const bf16x8*)(wc + 16);
            v16bf bfrag;
#pragma unroll
            for (int e = 0; e < 8; ++e) { bfrag[e] = b0[e]; bfrag[8 + e] = b1[e]; }
            acc[nt] = __builtin_amdgcn_wmma_f32_16x16x32_bf16(
                false, afrag, false, bfrag, (short)0, acc[nt], false, false);
        }
    }

    // bias + scale + bf16 store (C layout: row = v + 8*lhalf, col = nt*16+lmod)
#pragma unroll
    for (int nt = 0; nt < 4; ++nt) {
        float bv = bias[nt * 16 + lmod];
#pragma unroll
        for (int v = 0; v < 8; ++v) {
            long row = rowBase + v + lhalf * 8;
            out[row * H_DIM + nt * 16 + lmod] = (__bf16)((acc[nt][v] + bv) * scale);
        }
    }
}

// ============================================================
// Kernel 2: flash attention over bf16 q/k/v, fp32 out.
// Block = 128 threads = 4 waves; wave owns 16 query rows.
// K tile: DMA'd into LDS by the Tensor Data Mover (row-major fits the
// scores B-fragment layout directly).  V tile: manually staged transposed
// so the PV B-fragment is contiguous.  q is pre-scaled by log2(e)/8 so
// softmax runs in exp2 domain (one v_exp_f32 per probability).
// ============================================================
__global__ __launch_bounds__(128) void attn_kernel(
    const __bf16* __restrict__ qp, const __bf16* __restrict__ kp,
    const __bf16* __restrict__ vp, float* __restrict__ out)
{
    __shared__ __align__(16) __bf16 ktile[32 * H_DIM];   // [key][h]  4 KB (TDM dest)
    __shared__ __align__(16) __bf16 vtileT[H_DIM * 32];  // [h][key]  4 KB
    __shared__ __align__(16) __bf16 ptile[4][16 * 32];   // per-wave P 4 KB

    const int b     = blockIdx.y;
    const int tid   = threadIdx.x;
    const int wave  = tid >> 5;
    const int lane  = tid & 31;
    const int lhalf = lane >> 4;
    const int lmod  = lane & 15;
    const long qbase = (long)b * S_DIM + blockIdx.x * 64 + wave * 16;

    const __bf16* kb = kp + (long)b * S_DIM * H_DIM;
    const __bf16* vb = vp + (long)b * S_DIM * H_DIM;

    // Q A-fragments for the two K=32 steps covering H=64 (q pre-scaled)
    v16bf qa[2];
    const __bf16* qrow = qp + (qbase + lmod) * H_DIM;
#pragma unroll
    for (int kk = 0; kk < 2; ++kk) {
        bf16x8 r0 = *(const bf16x8*)(qrow + kk * 32 + lhalf * 8);
        bf16x8 r1 = *(const bf16x8*)(qrow + kk * 32 + 16 + lhalf * 8);
#pragma unroll
        for (int e = 0; e < 8; ++e) { qa[kk][e] = r0[e]; qa[kk][8 + e] = r1[e]; }
    }

    float run_m[8], run_l[8];
#pragma unroll
    for (int v = 0; v < 8; ++v) { run_m[v] = -INFINITY; run_l[v] = 0.0f; }
    v8f oacc[4] = {};

    const unsigned ldsK = (unsigned)(size_t)(void*)&ktile[0]; // LDS aperture: low 32 bits = offset

    for (int key0 = 0; key0 < S_DIM; key0 += 32) {
        __syncthreads();   // prior iteration done reading ktile/vtileT

        // ---- TDM: async-load K tile [32 x 64] bf16 into LDS (wave 0) ----
        if (wave == 0) {
            unsigned long long ga =
                (unsigned long long)(size_t)(const void*)(kb + (long)key0 * H_DIM);
            u32x4 g0 = { 1u,                                  // count=1, user desc
                         ldsK,                                // lds_addr
                         (unsigned)(ga & 0xFFFFFFFFu),        // global_addr[31:0]
                         (((unsigned)(ga >> 32)) & 0x01FFFFFFu) | 0x80000000u }; // addr[56:32] | type=2
            i32x8 g1 = { 0x00010000,                          // wg_mask=0, data_size=2B
                         (int)(64u << 16),                    // tensor_dim0 = 64
                         (int)(2048u << 16),                  // tensor_dim1 = 2048 (lo16)
                         (int)(64u << 16),                    // tile_dim0 = 64
                         32,                                  // tile_dim1 = 32, tile_dim2 = 0
                         64,                                  // tensor_dim0_stride = 64
                         0,                                   // stride0 hi / stride1 lo
                         2 };                                 // tensor_dim1_stride = 131072
            i32x4 gz4 = { 0, 0, 0, 0 };
            i32x8 gz8 = { 0, 0, 0, 0, 0, 0, 0, 0 };
            __builtin_amdgcn_tensor_load_to_lds(g0, g1, gz4, gz4, gz8, 0);
        }

        // ---- stage V tile transposed->LDS (overlaps TDM) ----
#pragma unroll
        for (int t = 0; t < 8; ++t) {
            int p  = tid + 128 * t;          // 0..1023
            int h  = p & 63;
            int kp2 = p >> 6;                // key pair index 0..15
            bf16x2 pk;
            pk[0] = vb[(long)(key0 + 2 * kp2) * H_DIM + h];
            pk[1] = vb[(long)(key0 + 2 * kp2 + 1) * H_DIM + h];
            *(bf16x2*)&vtileT[h * 32 + 2 * kp2] = pk;
        }

        if (wave == 0) __builtin_amdgcn_s_wait_tensorcnt(0);
        __syncthreads();

        // ---- scores: S = Q @ K^T for 32 keys (2 N-tiles x 2 K-steps) ----
        v8f sacc[2] = {};
#pragma unroll
        for (int nt = 0; nt < 2; ++nt) {
#pragma unroll
            for (int kk = 0; kk < 2; ++kk) {
                // B fragment: col = key, k = h = kk*32 + runs [lhalf*8..+8) and [16+lhalf*8..+8)
                const __bf16* krow = &ktile[(nt * 16 + lmod) * H_DIM + kk * 32 + lhalf * 8];
                bf16x8 c0 = *(const bf16x8*)(krow);
                bf16x8 c1 = *(const bf16x8*)(krow + 16);
                v16bf bfrag;
#pragma unroll
                for (int e = 0; e < 8; ++e) { bfrag[e] = c0[e]; bfrag[8 + e] = c1[e]; }
                sacc[nt] = __builtin_amdgcn_wmma_f32_16x16x32_bf16(
                    false, qa[kk], false, bfrag, (short)0, sacc[nt], false, false);
            }
        }

        // ---- online softmax in exp2 domain ----
        float p0a[8], p1a[8];
#pragma unroll
        for (int v = 0; v < 8; ++v) {
            float s0 = sacc[0][v];
            float s1 = sacc[1][v];
            float m = fmaxf(s0, s1);
#pragma unroll
            for (int off = 1; off < 16; off <<= 1)
                m = fmaxf(m, __shfl_xor(m, off));
            float nm   = fmaxf(run_m[v], m);
            float corr = exp2f(run_m[v] - nm);
            float p0 = exp2f(s0 - nm);
            float p1 = exp2f(s1 - nm);
            float rs = p0 + p1;
#pragma unroll
            for (int off = 1; off < 16; off <<= 1)
                rs += __shfl_xor(rs, off);
            run_l[v] = run_l[v] * corr + rs;
            run_m[v] = nm;
#pragma unroll
            for (int nt2 = 0; nt2 < 4; ++nt2) oacc[nt2][v] *= corr;
            p0a[v] = p0; p1a[v] = p1;
        }

        // ---- C-layout P -> LDS -> A-layout fragment (per-wave scratch) ----
        __bf16* pw = &ptile[wave][0];
#pragma unroll
        for (int v = 0; v < 8; ++v) {
            int row = v + lhalf * 8;
            pw[row * 32 + lmod]      = (__bf16)p0a[v];
            pw[row * 32 + 16 + lmod] = (__bf16)p1a[v];
        }
        v16bf pa;
        {
            const __bf16* prow = pw + lmod * 32 + lhalf * 8;
            bf16x8 r0 = *(const bf16x8*)(prow);
            bf16x8 r1 = *(const bf16x8*)(prow + 16);
#pragma unroll
            for (int e = 0; e < 8; ++e) { pa[e] = r0[e]; pa[8 + e] = r1[e]; }
        }

        // ---- O += P @ V : B fragment contiguous from vtileT ----
#pragma unroll
        for (int nt = 0; nt < 4; ++nt) {
            const __bf16* vc = &vtileT[(nt * 16 + lmod) * 32 + lhalf * 8];
            bf16x8 b0 = *(const bf16x8*)(vc);
            bf16x8 b1 = *(const bf16x8*)(vc + 16);
            v16bf bfrag;
#pragma unroll
            for (int e = 0; e < 8; ++e) { bfrag[e] = b0[e]; bfrag[8 + e] = b1[e]; }
            oacc[nt] = __builtin_amdgcn_wmma_f32_16x16x32_bf16(
                false, pa, false, bfrag, (short)0, oacc[nt], false, false);
        }
    }

    // ---- epilogue: normalize and store fp32 ----
#pragma unroll
    for (int v = 0; v < 8; ++v) {
        float inv = 1.0f / run_l[v];
        long row = qbase + v + lhalf * 8;
#pragma unroll
        for (int nt = 0; nt < 4; ++nt)
            out[row * H_DIM + nt * 16 + lmod] = oacc[nt][v] * inv;
    }
}

// ============================================================
extern "C" void kernel_launch(void* const* d_in, const int* in_sizes, int n_in,
                              void* d_out, int out_size, void* d_ws, size_t ws_size,
                              hipStream_t stream) {
    const float* query = (const float*)d_in[0];
    const float* key   = (const float*)d_in[1];
    const float* value = (const float*)d_in[2];
    const float* Wq    = (const float*)d_in[3];
    const float* bq    = (const float*)d_in[4];
    const float* Wk    = (const float*)d_in[5];
    const float* bk    = (const float*)d_in[6];
    const float* Wv    = (const float*)d_in[7];
    const float* bv    = (const float*)d_in[8];
    float* out = (float*)d_out;

    const size_t rows = (size_t)B_DIM * S_DIM;          // 16384
    __bf16* qp = (__bf16*)d_ws;
    __bf16* kp = qp + rows * H_DIM;
    __bf16* vp = kp + rows * H_DIM;

    // fold 1/sqrt(H) * log2(e) into q so softmax runs in exp2 domain
    const float qscale = 0.125f * 1.44269504088896340736f;

    dim3 blk(128);
    dim3 projGrid((unsigned)(rows / 64));               // 256 blocks
    proj_kernel<<<projGrid, blk, 0, stream>>>(query, Wq, bq, qscale, qp);
    proj_kernel<<<projGrid, blk, 0, stream>>>(key,   Wk, bk, 1.0f,  kp);
    proj_kernel<<<projGrid, blk, 0, stream>>>(value, Wv, bv, 1.0f,  vp);

    dim3 attnGrid(S_DIM / 64, B_DIM);                   // 32 x 8
    attn_kernel<<<attnGrid, blk, 0, stream>>>(qp, kp, vp, out);
}
